// LSTM_14577119002876
// MI455X (gfx1250) — matrix operand order repaired and checked by reading, not compiled, not run on hardware
//
#include <hip/hip_runtime.h>
#include <hip/hip_bf16.h>

typedef _Float16 half8  __attribute__((ext_vector_type(8)));
typedef _Float16 half16 __attribute__((ext_vector_type(16)));
typedef float    float8 __attribute__((ext_vector_type(8)));

#define B_   128
#define T_   512
#define IN_  64
#define H_   512
#define H2_  256
#define G4_  2048   // 4*H
#define NBLK 32     // j-tile blocks: 32 * 16 = 512 hidden units
#define JT   16

#define LDS_W_BYTES  ((size_t)(64 * IN_ + 3 * 64 * H_) * 2)   // 204800 (256B aligned)
#define LDS_G_FLOATS (4 * 8 * 32 * 8)                          // one gate-exchange buf: 8192 f32 = 32KB

// ---------------- math helpers ----------------
__device__ __forceinline__ float sigf(float x)       { return 1.0f / (1.0f + __expf(-x)); }
__device__ __forceinline__ float tanhf_fast(float x) { return 2.0f / (1.0f + __expf(-2.0f * x)) - 1.0f; }

__device__ __forceinline__ half16 mk16(half8 x, half8 y) {
  return __builtin_shufflevector(x, y, 0,1,2,3,4,5,6,7,8,9,10,11,12,13,14,15);
}

// ---------------- WMMA fragment loads (wave32 layouts per CDNA5 ISA 7.12.2) ----
// A (16x32 f16, M x K): lane&15 = M row, lane>>4 picks K-half.
__device__ __forceinline__ half16 load_a(const _Float16* __restrict__ A, int lda,
                                         int m0, int k0, int lane) {
  int m  = m0 + (lane & 15);
  int kh = (lane >> 4) * 8;
  const _Float16* row = A + (size_t)m * lda + k0;
  half8 x = *(const half8*)(row + kh);
  half8 y = *(const half8*)(row + 16 + kh);
  return mk16(x, y);
}

// B (32x16 f16, K x N) from LDS weight tile (64 rows x rowlen), row = g*16 + n.
__device__ __forceinline__ half16 load_b_lds(const _Float16* __restrict__ tile, int rowlen,
                                             int g, int k0, int lane) {
  int r = g * 16 + (lane & 15);
  int k = k0 + (lane >> 4) * 8;
  const half8* p = (const half8*)(tile + (size_t)r * rowlen + k);
  return mk16(p[0], p[2]);
}

__device__ __forceinline__ float8 wmma_f16(half16 a, half16 b, float8 c) {
  return __builtin_amdgcn_wmma_f32_16x16x32_f16(false, a, false, b, (short)0, c, false, false);
}

// Copy this block's 64 gate rows (4 gates x 16 units at j0) of W (4H x rowlen) into LDS.
__device__ void stage_tile(_Float16* __restrict__ dst, const _Float16* __restrict__ src,
                           int j0, int rowlen) {
  int vr = rowlen >> 3;  // half8 vectors per row
  for (int i = threadIdx.x; i < 64 * vr; i += blockDim.x) {
    int r = i / vr, v = (i - r * vr) << 3;
    int n = (r >> 4) * H_ + j0 + (r & 15);
    *(half8*)(dst + (size_t)r * rowlen + v) = *(const half8*)(src + (size_t)n * rowlen + v);
  }
}

// ---------------- device-wide barrier (bar[0]=count, bar[1]=generation) -------
// Writers must __threadfence() (release) before entering. On exit, EVERY thread
// performs an agent-scope acquire fence so per-CU L0s drop stale h-buffer lines.
__device__ __forceinline__ void grid_barrier(unsigned* __restrict__ bar, unsigned nb) {
  __syncthreads();
  if (threadIdx.x == 0) {
    volatile unsigned* gen = bar + 1;
    unsigned g = *gen;
    __threadfence();
    if (atomicAdd(bar, 1u) == nb - 1u) {
      atomicExch(bar, 0u);
      __threadfence();
      atomicAdd((unsigned*)gen, 1u);
    } else {
      while (*gen == g) { __builtin_amdgcn_s_sleep(1); }
    }
  }
  __syncthreads();
  __builtin_amdgcn_fence(__ATOMIC_ACQUIRE, "agent");   // global_inv on all waves
}

// ---------------- prep kernels ----------------
__global__ void f2h_kernel(const float* __restrict__ s, _Float16* __restrict__ d, int n) {
  int i = blockIdx.x * blockDim.x + threadIdx.x;
  if (i < n) d[i] = (_Float16)s[i];
}
__global__ void bias_kernel(const float* __restrict__ a, const float* __restrict__ b,
                            float* __restrict__ o, int n) {
  int i = blockIdx.x * blockDim.x + threadIdx.x;
  if (i < n) o[i] = a[i] + b[i];
}
// x (B,T,IN) f32 -> xT (T,B,IN) f16 for contiguous per-step A-operand loads
__global__ void xpose_kernel(const float* __restrict__ x, _Float16* __restrict__ xT) {
  int idx = blockIdx.x * blockDim.x + threadIdx.x;
  if (idx >= B_ * T_ * IN_) return;
  int i = idx % IN_;
  int t = (idx / IN_) % T_;
  int b = idx / (IN_ * T_);
  xT[((size_t)t * B_ + b) * IN_ + i] = (_Float16)x[idx];
}

// ---------------- fused 2-layer persistent LSTM -------------------------------
// 32 blocks x 1024 threads. Wave w: m-tile = w&7 (16 batch rows), gate = w>>3.
// Each wave computes ONE gate's 16x16 accumulator; gates meet in LDS, then wave
// group g performs the pointwise update for rows {2g, 2g+1} of its m-tile,
// keeping that slice of c0/c1 in a register float2.
__global__ __launch_bounds__(1024, 1)
void lstm2_persistent(const _Float16* __restrict__ xT,
                      const _Float16* __restrict__ Wih0, const _Float16* __restrict__ Whh0,
                      const _Float16* __restrict__ Wih1, const _Float16* __restrict__ Whh1,
                      const float* __restrict__ bias0, const float* __restrict__ bias1,
                      _Float16* __restrict__ h0buf, _Float16* __restrict__ h1buf,
                      unsigned* __restrict__ bar) {
  extern __shared__ char smem_raw[];
  _Float16* sWih0 = (_Float16*)smem_raw;           // 64 x 64
  _Float16* sWhh0 = sWih0 + 64 * IN_;              // 64 x 512
  _Float16* sWih1 = sWhh0 + 64 * H_;               // 64 x 512
  _Float16* sWhh1 = sWih1 + 64 * H_;               // 64 x 512  (200 KB)
  float* sG_A = (float*)(smem_raw + LDS_W_BYTES);  // 32 KB gate exchange, phase A
  float* sG_B = sG_A + LDS_G_FLOATS;               // 32 KB gate exchange, phase B

  const int lane = threadIdx.x & 31;
  const int wave = threadIdx.x >> 5;   // 0..31
  const int mt   = wave & 7;           // m-tile
  const int g    = wave >> 3;          // gate (0=i, 1=f, 2=g, 3=o)
  const int j0   = blockIdx.x * JT;    // hidden-unit tile this block owns
  const int m0   = mt * 16;
  const int l15  = lane & 15;
  const int kh   = lane >> 4;

  // weights -> LDS, once for all 512 steps
  stage_tile(sWih0, Wih0, j0, IN_);
  stage_tile(sWhh0, Whh0, j0, H_);
  stage_tile(sWih1, Wih1, j0, H_);
  stage_tile(sWhh1, Whh1, j0, H_);
  __syncthreads();

  const float b0 = bias0[g * H_ + j0 + l15];
  const float b1 = bias1[g * H_ + j0 + l15];

  // this wave's slice of cell state: rows {2g, 2g+1} of m-tile mt
  float2 c0 = make_float2(0.f, 0.f);
  float2 c1 = make_float2(0.f, 0.f);

  // fragment-order slot for this wave's accumulator in the exchange buffers
  float* slotA = sG_A + (size_t)((g * 8 + mt) * 32 + lane) * 8;
  float* slotB = sG_B + (size_t)((g * 8 + mt) * 32 + lane) * 8;
  // this wave's update-read base: rows 2g..2g+1, gate q at +q*2048
  const float* updA = sG_A + (size_t)(mt * 32 + lane) * 8 + 2 * g;
  const float* updB = sG_B + (size_t)(mt * 32 + lane) * 8 + 2 * g;
  const size_t hIdx0 = (size_t)(m0 + kh * 8 + 2 * g)     * H_ + j0 + l15;
  const size_t hIdx1 = (size_t)(m0 + kh * 8 + 2 * g + 1) * H_ + j0 + l15;

  for (int t = 0; t < T_; ++t) {
    const _Float16* h0r = h0buf + (size_t)(t & 1) * (B_ * H_);
    _Float16*       h0w = h0buf + (size_t)((t + 1) & 1) * (B_ * H_);
    const _Float16* h1r = h1buf + (size_t)(t & 1) * (B_ * H_);
    _Float16*       h1w = h1buf + (size_t)((t + 1) & 1) * (B_ * H_);
    const _Float16* xt  = xT + (size_t)t * (B_ * IN_);

    if (t + 1 < T_) __builtin_prefetch(xT + (size_t)(t + 1) * (B_ * IN_), 0, 0);

    // ========== phase A (layer 0): my gate = x_t@Wih0^T + h0@Whh0^T + b ==========
    // two independent accumulator chains (even/odd k-chunks)
    float8 acc0, acc1;
#pragma unroll
    for (int r = 0; r < 8; ++r) { acc0[r] = b0; acc1[r] = 0.f; }

#pragma unroll 2
    for (int k0 = 0; k0 < H_; k0 += 64) {
      half16 ae = load_a(h0r, H_, m0, k0, lane);
      acc0 = wmma_f16(ae, load_b_lds(sWhh0, H_, g, k0, lane), acc0);
      half16 ao = load_a(h0r, H_, m0, k0 + 32, lane);
      acc1 = wmma_f16(ao, load_b_lds(sWhh0, H_, g, k0 + 32, lane), acc1);
    }
    {
      half16 ax0 = load_a(xt, IN_, m0, 0, lane);
      acc0 = wmma_f16(ax0, load_b_lds(sWih0, IN_, g, 0, lane), acc0);
      half16 ax1 = load_a(xt, IN_, m0, 32, lane);
      acc1 = wmma_f16(ax1, load_b_lds(sWih0, IN_, g, 32, lane), acc1);
    }
    *(float8*)slotA = acc0 + acc1;
    __syncthreads();

    {   // distributed pointwise update: this wave does rows 2g, 2g+1
      float2 ai = *(const float2*)(updA + 0 * 2048);
      float2 af = *(const float2*)(updA + 1 * 2048);
      float2 ag = *(const float2*)(updA + 2 * 2048);
      float2 ao = *(const float2*)(updA + 3 * 2048);
      float cv0 = sigf(af.x) * c0.x + sigf(ai.x) * tanhf_fast(ag.x);
      float cv1 = sigf(af.y) * c0.y + sigf(ai.y) * tanhf_fast(ag.y);
      c0.x = cv0; c0.y = cv1;
      h0w[hIdx0] = (_Float16)(sigf(ao.x) * tanhf_fast(cv0));
      h0w[hIdx1] = (_Float16)(sigf(ao.y) * tanhf_fast(cv1));
      __threadfence();   // release before barrier arrival
    }
    grid_barrier(bar, NBLK);   // the ONLY device-wide barrier per timestep

    // ========== phase B (layer 1): my gate = h0_new@Wih1^T + h1@Whh1^T + b =======
#pragma unroll
    for (int r = 0; r < 8; ++r) { acc0[r] = b1; acc1[r] = 0.f; }

#pragma unroll 4
    for (int k0 = 0; k0 < H_; k0 += 32) {
      half16 a0 = load_a(h0w, H_, m0, k0, lane);
      acc0 = wmma_f16(a0, load_b_lds(sWih1, H_, g, k0, lane), acc0);
      half16 a1 = load_a(h1r, H_, m0, k0, lane);
      acc1 = wmma_f16(a1, load_b_lds(sWhh1, H_, g, k0, lane), acc1);
    }
    *(float8*)slotB = acc0 + acc1;
    __syncthreads();

    {
      float2 ai = *(const float2*)(updB + 0 * 2048);
      float2 af = *(const float2*)(updB + 1 * 2048);
      float2 ag = *(const float2*)(updB + 2 * 2048);
      float2 ao = *(const float2*)(updB + 3 * 2048);
      float cv0 = sigf(af.x) * c1.x + sigf(ai.x) * tanhf_fast(ag.x);
      float cv1 = sigf(af.y) * c1.y + sigf(ai.y) * tanhf_fast(ag.y);
      c1.x = cv0; c1.y = cv1;
      h1w[hIdx0] = (_Float16)(sigf(ao.x) * tanhf_fast(cv0));
      h1w[hIdx1] = (_Float16)(sigf(ao.y) * tanhf_fast(cv1));
      __threadfence();   // h1 writes become visible via next step's barrier
    }
    // no end-of-step grid barrier: every h0/h1 hazard is covered by the
    // phase-A barrier of this or the next step (ping-pong analysis), and the
    // phase-B __syncthreads covers LDS exchange-buffer reuse.
  }
}

// ---------------- FC head: out[b] = relu(h2_last @ fc1^T + b1) @ fc2^T + b2 ----
__global__ __launch_bounds__(256)
void fc_kernel(const _Float16* __restrict__ hlast,
               const float* __restrict__ w1, const float* __restrict__ b1,
               const float* __restrict__ w2, const float* __restrict__ b2,
               float* __restrict__ out) {
  __shared__ float red[H2_];
  int b = blockIdx.x, j = threadIdx.x;  // j = 0..255
  const _Float16* h = hlast + (size_t)b * H_;
  const float*    w = w1 + (size_t)j * H_;
  float acc = b1[j];
#pragma unroll 4
  for (int k = 0; k < H_; ++k) acc += (float)h[k] * w[k];
  red[j] = fmaxf(acc, 0.f) * w2[j];
  __syncthreads();
  for (int s = H2_ / 2; s > 0; s >>= 1) {
    if (j < s) red[j] += red[j + s];
    __syncthreads();
  }
  if (j == 0) out[b] = red[0] + b2[0];
}

// ---------------- host launcher ----------------
extern "C" void kernel_launch(void* const* d_in, const int* in_sizes, int n_in,
                              void* d_out, int out_size, void* d_ws, size_t ws_size,
                              hipStream_t stream) {
  (void)in_sizes; (void)n_in; (void)out_size; (void)ws_size;
  const float* x     = (const float*)d_in[0];
  const float* Wih0f = (const float*)d_in[1];
  const float* Whh0f = (const float*)d_in[2];
  const float* bih0  = (const float*)d_in[3];
  const float* bhh0  = (const float*)d_in[4];
  const float* Wih1f = (const float*)d_in[5];
  const float* Whh1f = (const float*)d_in[6];
  const float* bih1  = (const float*)d_in[7];
  const float* bhh1  = (const float*)d_in[8];
  const float* fc1w  = (const float*)d_in[9];
  const float* fc1b  = (const float*)d_in[10];
  const float* fc2w  = (const float*)d_in[11];
  const float* fc2b  = (const float*)d_in[12];
  float* out = (float*)d_out;

  char* ws = (char*)d_ws;
  size_t off = 0;
  auto alloc = [&](size_t bytes) {
    void* p = ws + off;
    off += (bytes + 255) & ~(size_t)255;
    return p;
  };
  _Float16* Wih0h = (_Float16*)alloc((size_t)G4_ * IN_ * 2);
  _Float16* Whh0h = (_Float16*)alloc((size_t)G4_ * H_ * 2);
  _Float16* Wih1h = (_Float16*)alloc((size_t)G4_ * H_ * 2);
  _Float16* Whh1h = (_Float16*)alloc((size_t)G4_ * H_ * 2);
  float*    bias0 = (float*)alloc((size_t)G4_ * 4);
  float*    bias1 = (float*)alloc((size_t)G4_ * 4);
  _Float16* xT    = (_Float16*)alloc((size_t)B_ * T_ * IN_ * 2);
  _Float16* h0buf = (_Float16*)alloc((size_t)2 * B_ * H_ * 2);
  _Float16* h1buf = (_Float16*)alloc((size_t)2 * B_ * H_ * 2);
  unsigned* bar   = (unsigned*)alloc(256);

  hipMemsetAsync(h0buf, 0, (size_t)2 * B_ * H_ * 2, stream);
  hipMemsetAsync(h1buf, 0, (size_t)2 * B_ * H_ * 2, stream);
  hipMemsetAsync(bar,   0, 256, stream);

  const int thr = 256;
  f2h_kernel<<<(G4_ * IN_ + thr - 1) / thr, thr, 0, stream>>>(Wih0f, Wih0h, G4_ * IN_);
  f2h_kernel<<<(G4_ * H_  + thr - 1) / thr, thr, 0, stream>>>(Whh0f, Whh0h, G4_ * H_);
  f2h_kernel<<<(G4_ * H_  + thr - 1) / thr, thr, 0, stream>>>(Wih1f, Wih1h, G4_ * H_);
  f2h_kernel<<<(G4_ * H_  + thr - 1) / thr, thr, 0, stream>>>(Whh1f, Whh1h, G4_ * H_);
  bias_kernel<<<(G4_ + thr - 1) / thr, thr, 0, stream>>>(bih0, bhh0, bias0, G4_);
  bias_kernel<<<(G4_ + thr - 1) / thr, thr, 0, stream>>>(bih1, bhh1, bias1, G4_);
  xpose_kernel<<<(B_ * T_ * IN_ + thr - 1) / thr, thr, 0, stream>>>(x, xT);

  const size_t ldsBytes = LDS_W_BYTES + 2 * (size_t)LDS_G_FLOATS * sizeof(float); // 264 KB
  hipFuncSetAttribute((const void*)lstm2_persistent,
                      hipFuncAttributeMaxDynamicSharedMemorySize, (int)ldsBytes);
  lstm2_persistent<<<NBLK, 1024, ldsBytes, stream>>>(xT, Wih0h, Whh0h, Wih1h, Whh1h,
                                                     bias0, bias1, h0buf, h1buf, bar);

  // final hidden state is in h1buf half 0 (T even => (T)&1 == 0)
  fc_kernel<<<B_, 256, 0, stream>>>(h1buf, fc1w, fc1b, fc2w, fc2b, out);
}
